// GroupedResidualVQ_65214783422724
// MI455X (gfx1250) — matrix operand order repaired and checked by reading, not compiled, not run on hardware
//
#include <hip/hip_runtime.h>

// ---------------------------------------------------------------------------
// Grouped Residual VQ forward for MI455X (gfx1250, wave32, WMMA).
// Distances via v_wmma_f32_16x16x32_bf16; argmin/skip/commit math exact f32.
// Round 2: use native __bf16 casts so the backend emits v_cvt_pk_bf16_f32
// instead of a 6-op integer RNE sequence per packed pair.
// ---------------------------------------------------------------------------

typedef __attribute__((ext_vector_type(16))) __bf16        v16bf;
typedef __attribute__((ext_vector_type(8)))  float         v8f;
typedef __attribute__((ext_vector_type(8)))  unsigned int  v8u;

#define NTOK   32768            // B*T = 16*2048
#define DDIM   768
#define NG     12
#define NM     3
#define NK     128
#define DGRP   64
#define SKIPID 128
#define CBP    36               // u32 pitch per codebook row (32 data + 4 pad -> 144B)
#define NROWS  (NM * NK)        // 384
#define WAVES  8
#define TPW    4                // 16-token tiles per wave
#define TOKS_PER_BLK (WAVES * TPW * 16)   // 512
#define NBLK_X (NTOK / TOKS_PER_BLK)      // 64
#define NBLK   (NBLK_X * NG)              // 768

static const size_t QOFF = (size_t)NTOK * DDIM;                 // start of indices
static const size_t SOFF = QOFF + (size_t)NTOK * (NG * NM);     // scalars

__device__ __forceinline__ unsigned int pack_bf16(float a, float b) {
  union { __bf16 h[2]; unsigned int u; } p;
  p.h[0] = (__bf16)a;           // hardware RNE convert (v_cvt_pk_bf16_f32)
  p.h[1] = (__bf16)b;
  return p.u;
}

__global__ __launch_bounds__(256) void grvq_main(
    const float* __restrict__ x, const float* __restrict__ cb,
    const float* __restrict__ lam, const int* __restrict__ valid,
    float* __restrict__ out, float* __restrict__ ws) {
  __shared__ unsigned int s_cb[NROWS * CBP];   // bf16-packed codebook rows
  __shared__ float        s_en[NROWS];         // exact f32 ||e||^2
  __shared__ int          s_idx[WAVES * 16];
  __shared__ float        s_red[3][WAVES];

  const int g    = blockIdx.y;
  const int tid  = threadIdx.x;
  const int lane = tid & 31;
  const int wave = tid >> 5;
  const int h    = lane >> 4;    // K-half of A/B fragments
  const int mrow = lane & 15;    // row (token) / column (code) within 16-tile

  // ---- stage this group's codebook into LDS (bf16) + exact norms ----------
  const float* cbg = cb + (size_t)g * (NM * NK * DGRP);
  for (int r = tid; r < NROWS; r += 256) {
    const float* src = cbg + (size_t)r * DGRP;
    unsigned int* dst = &s_cb[r * CBP];
    float nrm = 0.f;
#pragma unroll
    for (int j = 0; j < 32; ++j) {
      float a = src[2 * j], b = src[2 * j + 1];
      nrm += a * a + b * b;
      dst[j] = pack_bf16(a, b);
    }
    s_en[r] = nrm;
  }
  __syncthreads();

  const float thresh = lam[0] * 0.01f;
  float acc_commit = 0.f, acc_skip = 0.f, acc_valid = 0.f;

  for (int it = 0; it < TPW; ++it) {
    const int t0  = blockIdx.x * TOKS_PER_BLK + (it * WAVES + wave) * 16;
    const int tok = t0 + mrow;
    const int vmask  = valid[tok];
    const float validf = vmask ? 1.f : 0.f;

    // residual registers in A-fragment order:
    // chunk c (0..3) = dg [c*16 + 8h, c*16 + 8h + 8)
    float r[32], qs[32];
    const float* xr = x + (size_t)tok * DDIM + g * DGRP;
#pragma unroll
    for (int c = 0; c < 4; ++c) {
      const int off = c * 16 + 8 * h;
      float4 lo = *(const float4*)(xr + off);
      float4 hi = *(const float4*)(xr + off + 4);
      r[c * 8 + 0] = lo.x; r[c * 8 + 1] = lo.y; r[c * 8 + 2] = lo.z; r[c * 8 + 3] = lo.w;
      r[c * 8 + 4] = hi.x; r[c * 8 + 5] = hi.y; r[c * 8 + 6] = hi.z; r[c * 8 + 7] = hi.w;
    }
#pragma unroll
    for (int i = 0; i < 32; ++i) qs[i] = 0.f;

    for (int m = 0; m < NM; ++m) {
      // A fragments (bf16) for dg[0..31] and dg[32..63]
      v16bf a0, a1;
#pragma unroll
      for (int i = 0; i < 16; ++i) {
        a0[i] = (__bf16)r[i];
        a1[i] = (__bf16)r[16 + i];
      }

      float rn = 0.f;
#pragma unroll
      for (int i = 0; i < 32; ++i) rn += r[i] * r[i];
      rn += __shfl_xor(rn, 16, 32);               // exact ||r||^2 (d_skip)

      float best[8]; int bidx[8];
#pragma unroll
      for (int v = 0; v < 8; ++v) { best[v] = 3.0e38f; bidx[v] = 0; }

#pragma unroll
      for (int nt = 0; nt < 8; ++nt) {
        const int code = nt * 16 + mrow;
        const unsigned int* brow = &s_cb[(m * NK + code) * CBP];
        v8u b0p, b1p;
#pragma unroll
        for (int i = 0; i < 8; ++i) { b0p[i] = brow[4 * h + i + (i >= 4 ? 4 : 0)]; b1p[i] = brow[16 + 4 * h + i + (i >= 4 ? 4 : 0)]; }
        v16bf b0 = __builtin_bit_cast(v16bf, b0p);
        v16bf b1 = __builtin_bit_cast(v16bf, b1p);

        v8f c = {0.f, 0.f, 0.f, 0.f, 0.f, 0.f, 0.f, 0.f};
        c = __builtin_amdgcn_wmma_f32_16x16x32_bf16(false, a0, false, b0, (short)0, c, false, false);
        c = __builtin_amdgcn_wmma_f32_16x16x32_bf16(false, a1, false, b1, (short)0, c, false, false);

        const float en = s_en[m * NK + code];
#pragma unroll
        for (int v = 0; v < 8; ++v) {             // score = ||e||^2 - 2 r.e
          float s = en - 2.f * c[v];
          if (s < best[v]) { best[v] = s; bidx[v] = code; }
        }
      }

      // argmin across the 16 n-lanes (first-index tie-break)
#pragma unroll
      for (int off = 1; off < 16; off <<= 1) {
#pragma unroll
        for (int v = 0; v < 8; ++v) {
          float os = __shfl_xor(best[v], off, 32);
          int   oi = __shfl_xor(bidx[v], off, 32);
          if (os < best[v] || (os == best[v] && oi < bidx[v])) { best[v] = os; bidx[v] = oi; }
        }
      }
      if (mrow == 0) {                            // lanes 0 & 16 publish tokens v+8h
#pragma unroll
        for (int v = 0; v < 8; ++v) s_idx[wave * 16 + v + 8 * h] = bidx[v];
      }
      __builtin_amdgcn_wave_barrier();
      asm volatile("s_wait_dscnt 0" ::: "memory");
      const int midx = s_idx[wave * 16 + mrow];

      // exact f32 codeword gather (L2-resident)
      const float* qrow = cbg + ((size_t)(m * NK) + midx) * DGRP;
      float q[32];
#pragma unroll
      for (int c = 0; c < 4; ++c) {
        const int off = c * 16 + 8 * h;
        float4 lo = *(const float4*)(qrow + off);
        float4 hi = *(const float4*)(qrow + off + 4);
        q[c * 8 + 0] = lo.x; q[c * 8 + 1] = lo.y; q[c * 8 + 2] = lo.z; q[c * 8 + 3] = lo.w;
        q[c * 8 + 4] = hi.x; q[c * 8 + 5] = hi.y; q[c * 8 + 6] = hi.z; q[c * 8 + 7] = hi.w;
      }
      float ds = 0.f;
#pragma unroll
      for (int i = 0; i < 32; ++i) { float d = q[i] - r[i]; ds += d * d; }
      ds += __shfl_xor(ds, 16, 32);               // exact d_send = ||q - r||^2

      const float rel  = (rn - ds) / (rn + 1e-8f);
      const bool  skip = (rel < thresh) && (vmask != 0);

      if (lane < 16) {
        out[QOFF + (size_t)tok * (NG * NM) + g * NM + m] = (float)(skip ? SKIPID : midx);
        acc_commit += ds * validf;                // commit uses un-gated q
        acc_skip   += skip ? 1.f : 0.f;
      }
      if (!skip) {
#pragma unroll
        for (int i = 0; i < 32; ++i) { qs[i] += q[i]; r[i] -= q[i]; }
      }
    } // layers

    if (blockIdx.y == 0 && lane < 16) acc_valid += validf;

    float* orow = out + (size_t)tok * DDIM + g * DGRP;
#pragma unroll
    for (int c = 0; c < 4; ++c) {
      const int off = c * 16 + 8 * h;
      float4 lo = {qs[c * 8 + 0], qs[c * 8 + 1], qs[c * 8 + 2], qs[c * 8 + 3]};
      float4 hi = {qs[c * 8 + 4], qs[c * 8 + 5], qs[c * 8 + 6], qs[c * 8 + 7]};
      *(float4*)(orow + off)     = lo;
      *(float4*)(orow + off + 4) = hi;
    }
  } // tiles

  // deterministic per-block reduction (no atomics)
#pragma unroll
  for (int off = 16; off > 0; off >>= 1) {
    acc_commit += __shfl_xor(acc_commit, off, 32);
    acc_skip   += __shfl_xor(acc_skip,   off, 32);
    acc_valid  += __shfl_xor(acc_valid,  off, 32);
  }
  if (lane == 0) { s_red[0][wave] = acc_commit; s_red[1][wave] = acc_skip; s_red[2][wave] = acc_valid; }
  __syncthreads();
  if (tid == 0) {
    float c = 0.f, s = 0.f, v = 0.f;
    for (int w = 0; w < WAVES; ++w) { c += s_red[0][w]; s += s_red[1][w]; v += s_red[2][w]; }
    const int bid = blockIdx.y * NBLK_X + blockIdx.x;
    ws[bid]        = c;
    ws[NBLK + bid] = s;
    if (blockIdx.y == 0) ws[2 * NBLK + blockIdx.x] = v;
  }
}

__global__ void grvq_finish(float* __restrict__ out, const float* __restrict__ ws) {
  if (threadIdx.x != 0 || blockIdx.x != 0) return;
  float cs = 0.f, ss = 0.f, vs = 0.f;
  for (int i = 0; i < NBLK; ++i) { cs += ws[i]; ss += ws[NBLK + i]; }
  for (int i = 0; i < NBLK_X; ++i) vs += ws[2 * NBLK + i];
  const float vc = fmaxf(vs, 1.0f);
  out[SOFF]     = cs / (vc * (float)DGRP) / (float)(NM * NG);        // COMMIT_W = 1
  const float total_codes = vs * (float)(NG * NM);
  out[SOFF + 1] = ss / fmaxf(total_codes, 1.0f);
}

extern "C" void kernel_launch(void* const* d_in, const int* in_sizes, int n_in,
                              void* d_out, int out_size, void* d_ws, size_t ws_size,
                              hipStream_t stream) {
  const float* x     = (const float*)d_in[0];
  const float* cb    = (const float*)d_in[1];
  const float* lam   = (const float*)d_in[2];
  const int*   valid = (const int*)d_in[3];
  float* out = (float*)d_out;
  float* ws  = (float*)d_ws;

  dim3 grid(NBLK_X, NG), blk(256);
  grvq_main<<<grid, blk, 0, stream>>>(x, cb, lam, valid, out, ws);
  grvq_finish<<<1, 1, 0, stream>>>(out, ws);
  (void)in_sizes; (void)n_in; (void)out_size; (void)ws_size;
}
